// MoE10DimDecoder_24550033064021
// MI455X (gfx1250) — matrix operand order repaired and checked by reading, not compiled, not run on hardware
//
#include <hip/hip_runtime.h>

// ---------------------------------------------------------------------------
// Problem dimensions (must match reference)
// ---------------------------------------------------------------------------
constexpr int nB  = 2;
constexpr int nT  = 2048;
constexpr int nV  = 32000;
constexpr int nD  = 768;
constexpr int nNH = 12;
constexpr int nNKV= 4;
constexpr int nL  = 6;
constexpr int nNC = 64;
constexpr int nCD = 128;
constexpr int nHD = 64;
constexpr int nDI = 1536;
constexpr int nTOPK = 3;
constexpr float GATE_STRENGTH = 0.001f;
constexpr float RMS_EPS = 1e-6f;

typedef __attribute__((ext_vector_type(16))) _Float16 v16h;
typedef __attribute__((ext_vector_type(8)))  _Float16 h8;
typedef __attribute__((ext_vector_type(8)))  float    v8f;
typedef __attribute__((ext_vector_type(4)))  float    f4;

#define CAT16(lo, hi) __builtin_shufflevector(lo, hi, 0,1,2,3,4,5,6,7,8,9,10,11,12,13,14,15)

__device__ __forceinline__ h8 pack8(f4 a, f4 b)
{
    h8 r;
    r[0] = (_Float16)a[0]; r[1] = (_Float16)a[1];
    r[2] = (_Float16)a[2]; r[3] = (_Float16)a[3];
    r[4] = (_Float16)b[0]; r[5] = (_Float16)b[1];
    r[6] = (_Float16)b[2]; r[7] = (_Float16)b[3];
    return r;
}

__device__ __forceinline__ v8f wmma32(v16h a, v16h b, v8f c)
{
    return __builtin_amdgcn_wmma_f32_16x16x32_f16(false, a, false, b,
                                                  (short)0, c, false, false);
}

// ---------------------------------------------------------------------------
// Generic WMMA GEMM:  C[M,N] (=|+=) alpha * A[M,K] x B[K,N]
//   fp32 global -> f16 LDS tiles -> v_wmma_f32_16x16x32_f16, f32 accumulate.
//   REQUIRES: M % 64 == 0, N % 128 == 0, K % 32 == 0 (true for every call:
//   M in {2048,4096}, N in {256,768,1536,32000}, K in {768,1536}) -> no guards.
//   - transB: B element (k,n) read as Bp[n*ldb+k] (tied heads x @ W^T);
//     this path does contiguous float4 loads per thread.
//   - sel/bstride: expert indirection  Bp = Bbase + sel[sel_idx]*bstride
//   - alpha_ptr: device-side scalar (MoE top-k weight)
// Tile: BM=64, BN=128, BK=32; 256 threads = 8 waves (2M x 4N); each wave owns
// a 32x32 patch -> 4 16x16x32 WMMA accumulators per K-step (4 wmma / barrier).
// LDS: A row-major [m][k], B transposed [n][k]; rows padded to 40 halves so
// every lane fragment is two contiguous 16B chunks (ds_load_b128) with
// reduced bank conflicts.
// ---------------------------------------------------------------------------
#define BM 64
#define BN 128
#define BK 32
#define SA 40
#define SBT 40

__global__ void __launch_bounds__(256)
gemm_wmma_kernel(float* __restrict__ C,
                 const float* __restrict__ A,
                 const float* __restrict__ Bbase,
                 int M, int N, int K,
                 int lda, int ldb, int ldc,
                 int transB, int acc,
                 const float* __restrict__ alpha_ptr, int alpha_idx,
                 const int* __restrict__ sel, int sel_idx, long long bstride)
{
    __shared__ __attribute__((aligned(16))) _Float16 As [BM * SA];
    __shared__ __attribute__((aligned(16))) _Float16 Bst[BN * SBT];

    const float* Bp = Bbase + (sel ? (long long)sel[sel_idx] : 0ll) * bstride;

    const int tid  = threadIdx.x;
    const int wave = tid >> 5;
    const int lane = tid & 31;
    const int hl   = lane >> 4;   // wave half: 0 -> lanes 0-15, 1 -> lanes 16-31
    const int l16  = lane & 15;
    const int wm   = wave >> 2;   // 0..1  (M)
    const int wn   = wave & 3;    // 0..3  (N)

    const int tm0 = blockIdx.y * BM;
    const int tn0 = blockIdx.x * BN;

    // staging assignment
    const int ar  = tid >> 2;          // 0..63   A row
    const int akb = (tid & 3) * 8;     // 0,8,16,24
    const int bn  = tid & 127;         // 0..127  B column (of tile)
    const int bkb = (tid >> 7) * 16;   // 0,16

    v8f c00 = {}, c01 = {}, c10 = {}, c11 = {};

    for (int k0 = 0; k0 < K; k0 += BK) {
        // ---- stage A tile: 8 contiguous floats/thread -> one ds_store_b128
        {
            const float* ap = A + (long long)(tm0 + ar) * lda + k0 + akb;
            const f4 x0 = *(const f4*)ap;
            const f4 x1 = *(const f4*)(ap + 4);
            *(h8*)&As[ar * SA + akb] = pack8(x0, x1);
        }
        // ---- stage B tile transposed [n][k]
        if (transB) {
            // element (k,n) = Bp[n*ldb + k]: 16 contiguous floats per thread
            const float* bp = Bp + (long long)(tn0 + bn) * ldb + k0 + bkb;
            const f4 x0 = *(const f4*)bp;
            const f4 x1 = *(const f4*)(bp + 4);
            const f4 x2 = *(const f4*)(bp + 8);
            const f4 x3 = *(const f4*)(bp + 12);
            *(h8*)&Bst[bn * SBT + bkb]     = pack8(x0, x1);
            *(h8*)&Bst[bn * SBT + bkb + 8] = pack8(x2, x3);
        } else {
            // k-strided per thread, coalesced across the 128 consecutive n
            const float* bp = Bp + (long long)(k0 + bkb) * ldb + tn0 + bn;
            h8 r0, r1;
            #pragma unroll
            for (int j = 0; j < 8; ++j)
                r0[j] = (_Float16)bp[(long long)j * ldb];
            #pragma unroll
            for (int j = 0; j < 8; ++j)
                r1[j] = (_Float16)bp[(long long)(j + 8) * ldb];
            *(h8*)&Bst[bn * SBT + bkb]     = r0;
            *(h8*)&Bst[bn * SBT + bkb + 8] = r1;
        }
        // ---- prefetch next K tile (global_prefetch_b8)
        if (k0 + BK < K) {
            __builtin_prefetch(A + (long long)(tm0 + ar) * lda + k0 + BK + akb, 0, 1);
            if (transB)
                __builtin_prefetch(Bp + (long long)(tn0 + bn) * ldb + k0 + BK + bkb, 0, 1);
            else
                __builtin_prefetch(Bp + (long long)(k0 + BK + bkb) * ldb + tn0 + bn, 0, 1);
        }
        __syncthreads();

        // ---- fragments (contiguous k = 16*hl + s, same for A and B) ------------------------
        // A 16x32: lanes 0-15 row=lane K=0..15; lanes 16-31 K=16..31
        v16h a0, a1, b0, b1;
        {
            const _Float16* p = &As[(wm * 32 + l16) * SA + hl * 16];
            a0 = CAT16(*(const h8*)p, *(const h8*)(p + 8));
        }
        {
            const _Float16* p = &As[(wm * 32 + 16 + l16) * SA + hl * 16];
            a1 = CAT16(*(const h8*)p, *(const h8*)(p + 8));
        }
        // B 32x16: lanes 0-15 col=lane K=0..15; lanes 16-31 col=lane-16 K=16..31
        {
            const _Float16* p = &Bst[(wn * 32 + l16) * SBT + hl * 16];
            b0 = CAT16(*(const h8*)p, *(const h8*)(p + 8));
        }
        {
            const _Float16* p = &Bst[(wn * 32 + 16 + l16) * SBT + hl * 16];
            b1 = CAT16(*(const h8*)p, *(const h8*)(p + 8));
        }

        c00 = wmma32(a0, b0, c00);
        c01 = wmma32(a0, b1, c01);
        c10 = wmma32(a1, b0, c10);
        c11 = wmma32(a1, b1, c11);
        __syncthreads();
    }

    const float alpha = alpha_ptr ? alpha_ptr[alpha_idx] : 1.0f;

    // C/D 16x16 f32: VGPR r -> rows r (lanes 0-15), r+8 (lanes 16-31), col=l16
    const int gc0 = tn0 + wn * 32 + l16;
    #pragma unroll
    for (int r = 0; r < 8; ++r) {
        const int gr0 = tm0 + wm * 32 + r + 8 * hl;
        const int gr1 = gr0 + 16;
        {
            const long long o = (long long)gr0 * ldc + gc0;
            const float v0 = alpha * c00[r];
            const float v1 = alpha * c01[r];
            C[o]      = acc ? C[o]      + v0 : v0;
            C[o + 16] = acc ? C[o + 16] + v1 : v1;
        }
        {
            const long long o = (long long)gr1 * ldc + gc0;
            const float v0 = alpha * c10[r];
            const float v1 = alpha * c11[r];
            C[o]      = acc ? C[o]      + v0 : v0;
            C[o + 16] = acc ? C[o + 16] + v1 : v1;
        }
    }
}

// ---------------------------------------------------------------------------
// Embedding gather: x[b,t,:] = tok_emb[idx[b,t],:]
// ---------------------------------------------------------------------------
__global__ void embed_kernel(float* __restrict__ x,
                             const int* __restrict__ idx,
                             const float* __restrict__ emb)
{
    const long long i = (long long)blockIdx.x * blockDim.x + threadIdx.x;
    if (i >= (long long)nB * nT * nD) return;
    const long long tok = i / nD;
    const int d = (int)(i % nD);
    x[i] = emb[(long long)idx[tok] * nD + d];
}

// ---------------------------------------------------------------------------
// RMSNorm per row: y = x * rsqrt(mean(x^2)+eps) * w
// ---------------------------------------------------------------------------
__global__ void __launch_bounds__(256)
rmsnorm_kernel(float* __restrict__ y, const float* __restrict__ x,
               const float* __restrict__ w, int ncols)
{
    __shared__ float red[256];
    const long long row = blockIdx.x;
    const float* xr = x + row * ncols;
    float s = 0.0f;
    for (int c = threadIdx.x; c < ncols; c += 256) {
        const float v = xr[c];
        s += v * v;
    }
    red[threadIdx.x] = s;
    __syncthreads();
    for (int off = 128; off > 0; off >>= 1) {
        if ((int)threadIdx.x < off) red[threadIdx.x] += red[threadIdx.x + off];
        __syncthreads();
    }
    const float inv = rsqrtf(red[0] / (float)ncols + RMS_EPS);
    float* yr = y + row * ncols;
    for (int c = threadIdx.x; c < ncols; c += 256)
        yr[c] = xr[c] * inv * w[c];
}

// ---------------------------------------------------------------------------
// SwiGLU pointwise: out = silu(g)*u ; optional tension[row] = tanh(mean(out))
// ---------------------------------------------------------------------------
__global__ void __launch_bounds__(256)
swiglu_kernel(float* __restrict__ out, const float* __restrict__ g,
              const float* __restrict__ u, float* __restrict__ tension,
              int ncols)
{
    __shared__ float red[256];
    const long long row = blockIdx.x;
    const float* gr = g + row * ncols;
    const float* ur = u + row * ncols;
    float* orow = out + row * ncols;
    float s = 0.0f;
    for (int c = threadIdx.x; c < ncols; c += 256) {
        const float gv = gr[c];
        const float hv = gv / (1.0f + __expf(-gv)) * ur[c];
        orow[c] = hv;
        s += hv;
    }
    if (tension) {
        red[threadIdx.x] = s;
        __syncthreads();
        for (int off = 128; off > 0; off >>= 1) {
            if ((int)threadIdx.x < off) red[threadIdx.x] += red[threadIdx.x + off];
            __syncthreads();
        }
        if (threadIdx.x == 0) tension[row] = tanhf(red[0] / (float)ncols);
    }
}

// ---------------------------------------------------------------------------
// Consciousness-feedback add: x += tension[token] * tension_w[d] * GATE_STRENGTH
// ---------------------------------------------------------------------------
__global__ void csadd_kernel(float* __restrict__ x,
                             const float* __restrict__ tension,
                             const float* __restrict__ tw)
{
    const long long i = (long long)blockIdx.x * blockDim.x + threadIdx.x;
    if (i >= (long long)nB * nT * nD) return;
    x[i] += tension[i / nD] * tw[i % nD] * GATE_STRENGTH;
}

// ---------------------------------------------------------------------------
// Consciousness vector: pooled = mean_c(states); cv = sigmoid(gelu(pooled@W1+b1)@W2+b2)
// ---------------------------------------------------------------------------
__global__ void cv_kernel(float* __restrict__ cv,
                          const float* __restrict__ cstate,
                          const float* __restrict__ w1, const float* __restrict__ b1,
                          const float* __restrict__ w2, const float* __restrict__ b2)
{
    __shared__ float pooled[nCD];
    __shared__ float h1[64];
    const int b = blockIdx.x;
    const int t = threadIdx.x;
    if (t < nCD) {
        float s = 0.0f;
        for (int c = 0; c < nNC; ++c)
            s += cstate[((long long)b * nNC + c) * nCD + t];
        pooled[t] = s / (float)nNC;
    }
    __syncthreads();
    if (t < 64) {
        float s = b1[t];
        for (int d = 0; d < nCD; ++d) s += pooled[d] * w1[d * 64 + t];
        const float x3 = s * s * s;  // tanh-approx gelu (jax default)
        h1[t] = 0.5f * s * (1.0f + tanhf(0.7978845608f * (s + 0.044715f * x3)));
    }
    __syncthreads();
    if (t < 10) {
        float s = b2[t];
        for (int d = 0; d < 64; ++d) s += h1[d] * w2[d * 10 + t];
        cv[b * 10 + t] = 1.0f / (1.0f + __expf(-s));
    }
}

// ---------------------------------------------------------------------------
// Router: probs = softmax(cv@Wr+br); top-3, normalized weights
// ---------------------------------------------------------------------------
__global__ void route_kernel(int* __restrict__ ti, float* __restrict__ tw,
                             const float* __restrict__ cv,
                             const float* __restrict__ rw,
                             const float* __restrict__ rb)
{
    const int b = threadIdx.x;
    if (b >= nB) return;
    float p[10];
    float mx = -1e30f;
    for (int j = 0; j < 10; ++j) {
        float s = rb[j];
        for (int i = 0; i < 10; ++i) s += cv[b * 10 + i] * rw[i * 10 + j];
        p[j] = s;
        mx = fmaxf(mx, s);
    }
    float den = 0.0f;
    for (int j = 0; j < 10; ++j) { p[j] = __expf(p[j] - mx); den += p[j]; }
    for (int j = 0; j < 10; ++j) p[j] /= den;
    bool used[10] = {false};
    int   si[nTOPK];
    float sv[nTOPK];
    float sum3 = 0.0f;
    for (int kk = 0; kk < nTOPK; ++kk) {
        int bi = 0; float bv = -1.0f;
        for (int j = 0; j < 10; ++j)
            if (!used[j] && p[j] > bv) { bv = p[j]; bi = j; }
        used[bi] = true; si[kk] = bi; sv[kk] = bv; sum3 += bv;
    }
    for (int kk = 0; kk < nTOPK; ++kk) {
        ti[b * nTOPK + kk] = si[kk];
        tw[b * nTOPK + kk] = sv[kk] / sum3;
    }
}

// ---------------------------------------------------------------------------
// Causal GQA attention, per-thread online softmax, float4 loads.
// q: (B*T, NH*HD), k/v: (B*T, NKV*HD), o: (B*T, NH*HD)
// ---------------------------------------------------------------------------
__global__ void attn_kernel(const float* __restrict__ q,
                            const float* __restrict__ k,
                            const float* __restrict__ v,
                            float* __restrict__ o)
{
    const long long gid = (long long)blockIdx.x * blockDim.x + threadIdx.x;
    if (gid >= (long long)nB * nNH * nT) return;
    const int b   = (int)(gid / ((long long)nNH * nT));
    const int rem = (int)(gid % ((long long)nNH * nT));
    const int h   = rem / nT;
    const int t   = rem % nT;
    const int kvh = h / (nNH / nNKV);

    const f4* qp = (const f4*)(q + ((long long)(b * nT + t)) * (nNH * nHD) + h * nHD);
    f4 qr[nHD / 4], oacc[nHD / 4];
    #pragma unroll
    for (int d = 0; d < nHD / 4; ++d) {
        const f4 qv = qp[d];
        qr[d] = qv * 0.125f;
        oacc[d][0] = 0.0f; oacc[d][1] = 0.0f; oacc[d][2] = 0.0f; oacc[d][3] = 0.0f;
    }

    float m = -1e30f, lsum = 0.0f;
    for (int s = 0; s <= t; ++s) {
        const f4* kp = (const f4*)(k + ((long long)(b * nT + s)) * (nNKV * nHD) + kvh * nHD);
        float sc = 0.0f;
        #pragma unroll
        for (int d = 0; d < nHD / 4; ++d) {
            const f4 kv = kp[d];
            sc += qr[d][0] * kv[0] + qr[d][1] * kv[1] + qr[d][2] * kv[2] + qr[d][3] * kv[3];
        }
        const float nm    = fmaxf(m, sc);
        const float scale = __expf(m - nm);
        const float pp    = __expf(sc - nm);
        lsum = lsum * scale + pp;
        const f4* vp = (const f4*)(v + ((long long)(b * nT + s)) * (nNKV * nHD) + kvh * nHD);
        #pragma unroll
        for (int d = 0; d < nHD / 4; ++d)
            oacc[d] = oacc[d] * scale + vp[d] * pp;
        m = nm;
    }
    const float inv = 1.0f / lsum;
    f4* op = (f4*)(o + ((long long)(b * nT + t)) * (nNH * nHD) + h * nHD);
    #pragma unroll
    for (int d = 0; d < nHD / 4; ++d) op[d] = oacc[d] * inv;
}

// ---------------------------------------------------------------------------
// Host launch sequence
// ---------------------------------------------------------------------------
extern "C" void kernel_launch(void* const* d_in, const int* in_sizes, int n_in,
                              void* d_out, int out_size, void* d_ws, size_t ws_size,
                              hipStream_t stream)
{
    (void)in_sizes; (void)n_in; (void)out_size; (void)ws_size;

    const int*   idx       = (const int*)  d_in[0];
    const float* cstate    = (const float*)d_in[1];
    const float* tok_emb   = (const float*)d_in[2];
    const float* head_g    = (const float*)d_in[3];
    const float* cv_w1     = (const float*)d_in[4];
    const float* cv_b1     = (const float*)d_in[5];
    const float* cv_w2     = (const float*)d_in[6];
    const float* cv_b2     = (const float*)d_in[7];
    const float* tension_w = (const float*)d_in[8];
    const float* ln_attn_w = (const float*)d_in[9];
    const float* ln_pf_w   = (const float*)d_in[10];
    const float* ln_moe_w  = (const float*)d_in[11];
    const float* ln_f_w    = (const float*)d_in[12];
    const float* wq        = (const float*)d_in[13];
    const float* wk        = (const float*)d_in[14];
    const float* wv        = (const float*)d_in[15];
    const float* wo        = (const float*)d_in[16];
    const float* pf_gate   = (const float*)d_in[17];
    const float* pf_up     = (const float*)d_in[18];
    const float* pf_down   = (const float*)d_in[19];
    const float* e_gate    = (const float*)d_in[20];
    const float* e_up      = (const float*)d_in[21];
    const float* e_down    = (const float*)d_in[22];
    const float* router_w  = (const float*)d_in[23];
    const float* router_b  = (const float*)d_in[24];

    float* out = (float*)d_out;

    const long long XSZ  = (long long)nB * nT * nD;            // 3,145,728
    const long long KVSZ = (long long)nB * nT * nNKV * nHD;    // 1,048,576
    const long long HSZ  = (long long)nB * nT * nDI;           // 6,291,456
    const long long TOK  = (long long)nB * nT;

    float* ws = (float*)d_ws;
    long long off = 0;
    float* x  = ws + off; off += XSZ;
    float* h  = ws + off; off += XSZ;
    float* qb = ws + off; off += XSZ;
    float* ob = ws + off; off += XSZ;
    float* kb = ws + off; off += KVSZ;
    float* vb = ws + off; off += KVSZ;
    float* g1 = ws + off; off += HSZ;
    float* g2 = ws + off; off += HSZ;
    float* g3 = ws + off; off += HSZ;
    float* cvb = ws + off; off += 32;
    float* twb = ws + off; off += 32;
    int*   tib = (int*)(ws + off); off += 32;

    float* tens_base = out + 2ll * nB * nT * nV;   // tensions (L,B,T)

    auto gemm = [&](float* C, const float* A, const float* Bm,
                    int M, int N, int K, int lda, int ldb, int ldc,
                    int transB, int acc,
                    const float* ap, int ai,
                    const int* sel, int si, long long bs)
    {
        dim3 grid((unsigned)(N / BN), (unsigned)(M / BM));
        gemm_wmma_kernel<<<grid, dim3(256), 0, stream>>>(
            C, A, Bm, M, N, K, lda, ldb, ldc, transB, acc, ap, ai, sel, si, bs);
    };

    // ---- embedding + consciousness vector --------------------------------
    embed_kernel<<<(unsigned)((XSZ + 255) / 256), 256, 0, stream>>>(x, idx, tok_emb);
    cv_kernel<<<nB, 128, 0, stream>>>(cvb, cstate, cv_w1, cv_b1, cv_w2, cv_b2);

    const int M = nB * nT;  // 4096 token rows

    for (int l = 0; l < nL; ++l) {
        const float* wq_l  = wq + (long long)l * nD * (nNH * nHD);
        const float* wk_l  = wk + (long long)l * nD * (nNKV * nHD);
        const float* wv_l  = wv + (long long)l * nD * (nNKV * nHD);
        const float* wo_l  = wo + (long long)l * (nNH * nHD) * nD;
        const float* pfg_l = pf_gate + (long long)l * nD * nDI;
        const float* pfu_l = pf_up   + (long long)l * nD * nDI;
        const float* pfd_l = pf_down + (long long)l * nDI * nD;
        const float* eg_l  = e_gate  + (long long)l * 10 * nD * nDI;
        const float* eu_l  = e_up    + (long long)l * 10 * nD * nDI;
        const float* ed_l  = e_down  + (long long)l * 10 * nDI * nD;

        // ---- attention block ---------------------------------------------
        rmsnorm_kernel<<<M, 256, 0, stream>>>(h, x, ln_attn_w + l * nD, nD);
        gemm(qb, h, wq_l, M, nNH * nHD, nD, nD, nNH * nHD, nNH * nHD, 0, 0,
             nullptr, 0, nullptr, 0, 0);
        gemm(kb, h, wk_l, M, nNKV * nHD, nD, nD, nNKV * nHD, nNKV * nHD, 0, 0,
             nullptr, 0, nullptr, 0, 0);
        gemm(vb, h, wv_l, M, nNKV * nHD, nD, nD, nNKV * nHD, nNKV * nHD, 0, 0,
             nullptr, 0, nullptr, 0, 0);
        {
            const long long rows = (long long)nB * nNH * nT;
            attn_kernel<<<(unsigned)((rows + 127) / 128), 128, 0, stream>>>(qb, kb, vb, ob);
        }
        gemm(x, ob, wo_l, M, nD, nNH * nHD, nNH * nHD, nD, nD, 0, /*acc=*/1,
             nullptr, 0, nullptr, 0, 0);

        // ---- PureField SwiGLU FFN + tension ------------------------------
        rmsnorm_kernel<<<M, 256, 0, stream>>>(h, x, ln_pf_w + l * nD, nD);
        gemm(g1, h, pfg_l, M, nDI, nD, nD, nDI, nDI, 0, 0, nullptr, 0, nullptr, 0, 0);
        gemm(g2, h, pfu_l, M, nDI, nD, nD, nDI, nDI, 0, 0, nullptr, 0, nullptr, 0, 0);
        swiglu_kernel<<<M, 256, 0, stream>>>(g3, g1, g2,
                                             tens_base + (long long)l * TOK, nDI);
        gemm(x, g3, pfd_l, M, nD, nDI, nDI, nD, nD, 0, /*acc=*/1,
             nullptr, 0, nullptr, 0, 0);

        // ---- consciousness feedback from previous layer ------------------
        if (l > 0) {
            csadd_kernel<<<(unsigned)((XSZ + 255) / 256), 256, 0, stream>>>(
                x, tens_base + (long long)(l - 1) * TOK, tension_w);
        }

        // ---- consciousness-routed top-3 MoE ------------------------------
        rmsnorm_kernel<<<M, 256, 0, stream>>>(h, x, ln_moe_w + l * nD, nD);
        route_kernel<<<1, 32, 0, stream>>>(tib, twb, cvb,
                                           router_w + l * 100, router_b + l * 10);
        for (int b = 0; b < nB; ++b) {
            const float* Ab = h + (long long)b * nT * nD;
            float*       Cx = x + (long long)b * nT * nD;
            for (int kk = 0; kk < nTOPK; ++kk) {
                const int si = b * nTOPK + kk;
                gemm(g1, Ab, eg_l, nT, nDI, nD, nD, nDI, nDI, 0, 0,
                     nullptr, 0, tib, si, (long long)nD * nDI);
                gemm(g2, Ab, eu_l, nT, nDI, nD, nD, nDI, nDI, 0, 0,
                     nullptr, 0, tib, si, (long long)nD * nDI);
                swiglu_kernel<<<nT, 256, 0, stream>>>(g3, g1, g2, nullptr, nDI);
                gemm(Cx, g3, ed_l, nT, nD, nDI, nDI, nD, nD, 0, /*acc=*/1,
                     twb, si, tib, si, (long long)nDI * nD);
            }
        }
    }

    // ---- final norm + tied heads (x @ W^T) -------------------------------
    rmsnorm_kernel<<<M, 256, 0, stream>>>(h, x, ln_f_w, nD);
    gemm(out, h, tok_emb, M, nV, nD, nD, nD, nV, /*transB=*/1, 0,
         nullptr, 0, nullptr, 0, 0);
    gemm(out + (long long)nB * nT * nV, h, head_g, M, nV, nD, nD, nD, nV,
         /*transB=*/1, 0, nullptr, 0, nullptr, 0, 0);
}
